// SMMModel_41506563948813
// MI455X (gfx1250) — matrix operand order repaired and checked by reading, not compiled, not run on hardware
//
#include <hip/hip_runtime.h>
#include <hip/hip_bf16.h>
#include <stdint.h>

#define BATCH   16
#define SEQ     4096
#define IN_DIM  64
#define OUT_DIM 64
#define NSTATE  512
#define CHUNK   64
#define NCHUNK  (SEQ / CHUNK)

typedef __bf16 bf16_t;
typedef bf16_t v16bf __attribute__((ext_vector_type(16)));
typedef float  v8f   __attribute__((ext_vector_type(8)));

union V16U { v16bf v; uint4 q[2]; };

// ---------------------------------------------------------------------------
// WMMA operand loaders (wave32, 16-bit operands in this code's contiguous k order, same for A and B)
//
// A-operand 16x32 (M rows x K): lane l -> row m = l&15, half = l>>4.
//   elements e=0..15 hold K = k0 + half*16 + e       (contiguous 32B run)
// B-operand 32x16 (K x N cols): lane l -> col n = n0 + (l&15), half = l>>4.
//   elements e=0..15 hold K = k0 + half*16 + e       (contiguous 32B run)
// C/D 16x16 f32: vgpr v, lane l -> M = v + (l>>4)*8, N = l&15.
// ---------------------------------------------------------------------------
__device__ __forceinline__ v16bf load_bop(const bf16_t* W, int ld, int n0, int lane, int k0) {
  const int n = lane & 15, half = lane >> 4;
  const bf16_t* p = W + (size_t)(n0 + n) * ld + k0 + half * 16;
  V16U u;
  u.q[0] = *(const uint4*)(p);
  u.q[1] = *(const uint4*)(p + 8);
  return u.v;
}

__device__ __forceinline__ v16bf load_aop(const bf16_t* X, int ld, int lane, int k0) {
  const int m = lane & 15, half = lane >> 4;
  const bf16_t* p = X + (size_t)m * ld + k0 + half * 16;
  V16U u;
  u.q[0] = *(const uint4*)(p);
  u.q[1] = *(const uint4*)(p + 8);
  return u.v;
}

__device__ __forceinline__ v8f wmma_bf16(v16bf a, v16bf b, v8f c) {
  return __builtin_amdgcn_wmma_f32_16x16x32_bf16(false, a, false, b, (short)0, c, false, false);
}

// ---------------------------------------------------------------------------
// f32 -> bf16 casts
// ---------------------------------------------------------------------------
__global__ void k_cast_bf16(const float* __restrict__ s, bf16_t* __restrict__ d, int n) {
  int i = blockIdx.x * blockDim.x + threadIdx.x;
  int stride = gridDim.x * blockDim.x;
  for (; i < n; i += stride) d[i] = (bf16_t)s[i];
}

__global__ void k_transpose_bf16(const float* __restrict__ s, bf16_t* __restrict__ d,
                                 int rows, int cols) {
  int i = blockIdx.x * blockDim.x + threadIdx.x;
  int n = rows * cols;
  int stride = gridDim.x * blockDim.x;
  for (; i < n; i += stride) {
    int r = i / cols, c = i % cols;
    d[(size_t)c * rows + r] = (bf16_t)s[i];
  }
}

// ---------------------------------------------------------------------------
// Matrix squaring: (Mo, Mot) = (Mi*Mi, (Mi*Mi)^T), 512x512 bf16, f32 accum.
// out[m][n] = sum_k Mi[m][k] * Mit[n][k]   (Mit is Mi transposed)
// One wave per 16x16 output tile, K=512 -> 16 WMMAs.
// ---------------------------------------------------------------------------
__global__ __launch_bounds__(256) void k_sq(const bf16_t* __restrict__ Mi,
                                            const bf16_t* __restrict__ Mit,
                                            bf16_t* __restrict__ Mo,
                                            bf16_t* __restrict__ Mot) {
  const int tid  = threadIdx.x;
  const int lane = tid & 31;
  const int tile = blockIdx.x * (blockDim.x >> 5) + (tid >> 5); // 0..1023
  const int m0 = (tile >> 5) << 4;
  const int n0 = (tile & 31) << 4;
  v8f acc = {};
#pragma unroll
  for (int kt = 0; kt < 16; ++kt) {
    v16bf a = load_aop(Mi + (size_t)m0 * NSTATE, NSTATE, lane, kt * 32);
    v16bf b = load_bop(Mit, NSTATE, n0, lane, kt * 32);
    acc = wmma_bf16(a, b, acc);
  }
  const int half = lane >> 4, nn = n0 + (lane & 15);
#pragma unroll
  for (int v = 0; v < 8; ++v) {
    int m = m0 + v + half * 8;
    bf16_t e = (bf16_t)acc[v];
    Mo [(size_t)m  * NSTATE + nn] = e;
    Mot[(size_t)nn * NSTATE + m ] = e;
  }
}

// ---------------------------------------------------------------------------
// Unified scan kernel. One workgroup = 32 waves = full 512-wide state.
// Wave w owns state columns [w*16, w*16+16). Per step, per wave:
//   project mode: na  = WMMA(x_t[16x64], B^T) (2 WMMAs, x staged in LDS)
//   direct  mode: na  = rin[step]             (f32 loads)
//   then          na += WMMA(h[16x512], A^T)  (16 WMMAs, h in LDS as bf16)
// x_{t+1} is prefetched via global_load_async_to_lds_b128 (double-buffered).
// ---------------------------------------------------------------------------
#define HPAD 520   // 1040B rows -> 4-bank stagger, conflict-free 16B runs
#define XPAD 72    // 144B rows (16B aligned)

__global__ __launch_bounds__(1024) void k_scan(
    const bf16_t* __restrict__ W,    // [512][512] transition (row n, col m)
    const bf16_t* __restrict__ Bw,   // [512][64] input proj, or null => direct mode
    const bf16_t* __restrict__ xb,   // [16][4096][64] bf16 inputs (project mode)
    const float*  __restrict__ rin,  // [steps][16][512] additive term (direct mode)
    const float*  __restrict__ S0,   // [grid][16][512] initial states, or null => 0
    float*        __restrict__ rout, // [grid][16][512] final states, or null
    float*        __restrict__ Sout, // [steps][16][512] pre-update states, or null
    bf16_t*       __restrict__ hs,   // [16][4096][512] per-step states, or null
    int steps)
{
  __shared__ __align__(16) bf16_t lds_h[BATCH * HPAD];
  __shared__ __align__(16) bf16_t lds_x[2][BATCH * XPAD];

  const int tid  = threadIdx.x;
  const int lane = tid & 31;
  const int wave = tid >> 5;
  const int n0   = wave * 16;
  const int half = lane >> 4;
  const int nn   = n0 + (lane & 15);
  const int tbase = blockIdx.x * steps;
  const bool project = (Bw != nullptr);

  // Stationary input-projection tiles (K=64 -> 2 tiles).
  v16bf Breg0 = {}, Breg1 = {};
  if (project) {
    Breg0 = load_bop(Bw, IN_DIM, n0, lane, 0);
    Breg1 = load_bop(Bw, IN_DIM, n0, lane, 32);
  }

  // Initial state (f32 accumulator), mirrored to LDS as bf16.
  v8f acc = {};
  if (S0) {
    const float* s = S0 + (size_t)blockIdx.x * BATCH * NSTATE;
#pragma unroll
    for (int v = 0; v < 8; ++v)
      acc[v] = s[(size_t)(v + half * 8) * NSTATE + nn];
  }
#pragma unroll
  for (int v = 0; v < 8; ++v)
    lds_h[(v + half * 8) * HPAD + nn] = (bf16_t)acc[v];

  // Stage x_{t0} into LDS buffer 0 (async copy: 128 lanes x 16B = 2KB).
  if (project && tid < 128) {
    int b = tid >> 3, ch = tid & 7;
    const bf16_t* g = xb + ((size_t)b * SEQ + tbase) * IN_DIM + ch * 8;
    unsigned lo = (unsigned)(uintptr_t)&lds_x[0][b * XPAD + ch * 8];
    asm volatile("global_load_async_to_lds_b128 %0, %1, off"
                 :: "v"(lo), "v"((unsigned long long)(uintptr_t)g) : "memory");
    asm volatile("s_wait_asynccnt 0" ::: "memory");
  }
  __syncthreads();

  for (int s = 0; s < steps; ++s) {
    const int t   = tbase + s;
    const int buf = s & 1;

    // Pre-update state out (phase 2: chunk-entry states S_c).
    if (Sout) {
      float* o = Sout + (size_t)s * BATCH * NSTATE;
#pragma unroll
      for (int v = 0; v < 8; ++v)
        o[(size_t)(v + half * 8) * NSTATE + nn] = acc[v];
    }

    v8f na = {};
    if (project) {
      na = wmma_bf16(load_aop(lds_x[buf], XPAD, lane, 0),  Breg0, na);
      na = wmma_bf16(load_aop(lds_x[buf], XPAD, lane, 32), Breg1, na);
    } else {
      const float* rp = rin + (size_t)s * BATCH * NSTATE;
#pragma unroll
      for (int v = 0; v < 8; ++v)
        na[v] = rp[(size_t)(v + half * 8) * NSTATE + nn];
    }

    // h @ A^T : K=512 -> 16 WMMAs; A tiles stream from L2 (A is 512KB).
#pragma unroll
    for (int kt = 0; kt < 16; ++kt) {
      v16bf a = load_aop(lds_h, HPAD, lane, kt * 32);
      v16bf b = load_bop(W, NSTATE, n0, lane, kt * 32);
      na = wmma_bf16(a, b, na);
    }

    // Overlap: async-prefetch x_{t+1} into the other LDS buffer.
    if (project && tid < 128) {
      int tn = (t + 1 < SEQ) ? t + 1 : SEQ - 1;
      int b = tid >> 3, ch = tid & 7;
      const bf16_t* g = xb + ((size_t)b * SEQ + tn) * IN_DIM + ch * 8;
      unsigned lo = (unsigned)(uintptr_t)&lds_x[buf ^ 1][b * XPAD + ch * 8];
      asm volatile("global_load_async_to_lds_b128 %0, %1, off"
                   :: "v"(lo), "v"((unsigned long long)(uintptr_t)g) : "memory");
    }

    __syncthreads();  // all lds_h reads of this step complete

    acc = na;
#pragma unroll
    for (int v = 0; v < 8; ++v)
      lds_h[(v + half * 8) * HPAD + nn] = (bf16_t)acc[v];

    if (hs) {
#pragma unroll
      for (int v = 0; v < 8; ++v)
        hs[((size_t)(v + half * 8) * SEQ + t) * NSTATE + nn] = (bf16_t)acc[v];
    }

    if (project && tid < 128)
      asm volatile("s_wait_asynccnt 0" ::: "memory");

    __syncthreads();  // lds_h republished; x_{t+1} landed
  }

  if (rout) {
    float* o = rout + (size_t)blockIdx.x * BATCH * NSTATE;
#pragma unroll
    for (int v = 0; v < 8; ++v)
      o[(size_t)(v + half * 8) * NSTATE + nn] = acc[v];
  }
}

// ---------------------------------------------------------------------------
// y[b,t,o] = sum_n C[o,n] h[b,t,n] + sum_i D[o,i] x[b,t,i]
// Block = 4 waves; each wave: one 16(t) x 16(o) tile, K=512 (+64) -> 18 WMMAs.
// ---------------------------------------------------------------------------
__global__ __launch_bounds__(128) void k_ygemm(
    const bf16_t* __restrict__ hs, const bf16_t* __restrict__ xb,
    const bf16_t* __restrict__ Cw, const bf16_t* __restrict__ Dw,
    float* __restrict__ y)
{
  const int tid = threadIdx.x, lane = tid & 31, wave = tid >> 5;
  const int b  = blockIdx.x >> 8;          // 256 t-tiles per batch
  const int t0 = (blockIdx.x & 255) << 4;
  const int o0 = wave * 16;
  const int half = lane >> 4, oo = o0 + (lane & 15);

  v8f acc = {};
  const bf16_t* ha = hs + ((size_t)b * SEQ + t0) * NSTATE;
#pragma unroll
  for (int kt = 0; kt < 16; ++kt)
    acc = wmma_bf16(load_aop(ha, NSTATE, lane, kt * 32),
                    load_bop(Cw, NSTATE, o0, lane, kt * 32), acc);

  const bf16_t* xa = xb + ((size_t)b * SEQ + t0) * IN_DIM;
  acc = wmma_bf16(load_aop(xa, IN_DIM, lane, 0),  load_bop(Dw, IN_DIM, o0, lane, 0),  acc);
  acc = wmma_bf16(load_aop(xa, IN_DIM, lane, 32), load_bop(Dw, IN_DIM, o0, lane, 32), acc);

#pragma unroll
  for (int v = 0; v < 8; ++v) {
    int t = t0 + v + half * 8;
    y[((size_t)b * SEQ + t) * OUT_DIM + oo] = acc[v];
  }
}

// ---------------------------------------------------------------------------
// Host orchestration
// ---------------------------------------------------------------------------
extern "C" void kernel_launch(void* const* d_in, const int* in_sizes, int n_in,
                              void* d_out, int out_size, void* d_ws, size_t ws_size,
                              hipStream_t stream) {
  const float* x = (const float*)d_in[0];  // [16][4096][64]
  const float* A = (const float*)d_in[1];  // [512][512]
  const float* B = (const float*)d_in[2];  // [512][64]
  const float* C = (const float*)d_in[3];  // [64][512]
  const float* D = (const float*)d_in[4];  // [64][64]
  float* y = (float*)d_out;                // [16][4096][64]

  char* ws = (char*)d_ws;
  size_t off = 0;
  auto take = [&](size_t bytes) -> void* {
    void* p = ws + off;
    off = (off + bytes + 255) & ~(size_t)255;
    return p;
  };
  bf16_t* A_bf  = (bf16_t*)take((size_t)NSTATE * NSTATE * 2);
  bf16_t* A_bfT = (bf16_t*)take((size_t)NSTATE * NSTATE * 2);
  bf16_t* M1    = (bf16_t*)take((size_t)NSTATE * NSTATE * 2);
  bf16_t* M1t   = (bf16_t*)take((size_t)NSTATE * NSTATE * 2);
  bf16_t* M2    = (bf16_t*)take((size_t)NSTATE * NSTATE * 2);
  bf16_t* M2t   = (bf16_t*)take((size_t)NSTATE * NSTATE * 2);
  bf16_t* B_bf  = (bf16_t*)take((size_t)NSTATE * IN_DIM * 2);
  bf16_t* C_bf  = (bf16_t*)take((size_t)OUT_DIM * NSTATE * 2);
  bf16_t* D_bf  = (bf16_t*)take((size_t)OUT_DIM * IN_DIM * 2);
  bf16_t* xb    = (bf16_t*)take((size_t)BATCH * SEQ * IN_DIM * 2);
  float*  r     = (float*) take((size_t)NCHUNK * BATCH * NSTATE * 4);
  float*  S     = (float*) take((size_t)NCHUNK * BATCH * NSTATE * 4);
  bf16_t* hs    = (bf16_t*)take((size_t)BATCH * SEQ * NSTATE * 2);
  (void)ws_size; (void)in_sizes; (void)n_in; (void)out_size;

  // 1) bf16 conversions
  k_cast_bf16<<<256, 256, 0, stream>>>(A, A_bf, NSTATE * NSTATE);
  k_cast_bf16<<<64, 256, 0, stream>>>(B, B_bf, NSTATE * IN_DIM);
  k_cast_bf16<<<64, 256, 0, stream>>>(C, C_bf, OUT_DIM * NSTATE);
  k_cast_bf16<<<16, 256, 0, stream>>>(D, D_bf, OUT_DIM * IN_DIM);
  k_cast_bf16<<<1024, 256, 0, stream>>>(x, xb, BATCH * SEQ * IN_DIM);
  k_transpose_bf16<<<256, 256, 0, stream>>>(A, A_bfT, NSTATE, NSTATE);

  // 2) M = A^64 via 6 squarings (ping-pong, transposed twin maintained)
  k_sq<<<128, 256, 0, stream>>>(A_bf, A_bfT, M1, M1t);  // A^2
  k_sq<<<128, 256, 0, stream>>>(M1, M1t, M2, M2t);      // A^4
  k_sq<<<128, 256, 0, stream>>>(M2, M2t, M1, M1t);      // A^8
  k_sq<<<128, 256, 0, stream>>>(M1, M1t, M2, M2t);      // A^16
  k_sq<<<128, 256, 0, stream>>>(M2, M2t, M1, M1t);      // A^32
  k_sq<<<128, 256, 0, stream>>>(M1, M1t, M2, M2t);      // A^64 in M2

  // 3) Phase 1: per-chunk local scans (h0 = 0), keep final states r[c]
  k_scan<<<NCHUNK, 1024, 0, stream>>>(A_bf, B_bf, xb, nullptr, nullptr,
                                      r, nullptr, nullptr, CHUNK);
  // 4) Phase 2: boundary recurrence H_j = M H_{j-1} + r_j; emit entry states S
  k_scan<<<1, 1024, 0, stream>>>(M2, nullptr, nullptr, r, nullptr,
                                 nullptr, S, nullptr, NCHUNK);
  // 5) Phase 3: per-chunk scans from correct entry states, stream h_t to hs
  k_scan<<<NCHUNK, 1024, 0, stream>>>(A_bf, B_bf, xb, nullptr, S,
                                      nullptr, nullptr, hs, CHUNK);
  // 6) y = C h + D x
  k_ygemm<<<BATCH * (SEQ / 16), 128, 0, stream>>>(hs, xb, C_bf, D_bf, y);
}
